// AttentionwHook_764504179220
// MI455X (gfx1250) — matrix operand order repaired and checked by reading, not compiled, not run on hardware
//
#include <hip/hip_runtime.h>
#include <hip/hip_bf16.h>

typedef __attribute__((ext_vector_type(16))) _Float16 v16h;
typedef __attribute__((ext_vector_type(8)))  _Float16 v8h;
typedef __attribute__((ext_vector_type(8)))  float    v8f;
typedef __attribute__((ext_vector_type(4)))  float    v4f;

#define BATCH 8
#define SEQ   1024
#define CH    1024
#define HEADS 16
#define HDIM  64
#define SCALE 0.125f

// ---------------------------------------------------------------------------
// WMMA helpers (CDNA5 gfx1250, wave32)
// ---------------------------------------------------------------------------
__device__ __forceinline__ v8f wmma_f16(v16h a, v16h b, v8f c) {
  return __builtin_amdgcn_wmma_f32_16x16x32_f16(false, a, false, b, (short)0, c,
                                                false, false);
}

// A fragment (16x32 f16, M x K): lane m = l&15; lanes 0-15 hold
// K = {k0..k0+7, k0+16..k0+23}; lanes 16-31 shifted by +8.
__device__ __forceinline__ v16h load_fragA_h(const _Float16* __restrict__ p,
                                             int ld, int row0, int k0) {
  int lane = threadIdx.x & 31;
  const _Float16* base =
      p + (size_t)(row0 + (lane & 15)) * ld + k0 + ((lane >> 4) << 3);
  union { v16h v; v8h h[2]; } u;
  u.h[0] = *(const v8h*)(base);
  u.h[1] = *(const v8h*)(base + 16);
  return u.v;
}

// A fragment gathered from fp32 source (attn probabilities), converted in regs.
__device__ __forceinline__ v16h load_fragA_f32(const float* __restrict__ p,
                                               int ld, int row0, int k0) {
  int lane = threadIdx.x & 31;
  const float* base =
      p + (size_t)(row0 + (lane & 15)) * ld + k0 + ((lane >> 4) << 3);
  v4f a0 = *(const v4f*)(base);
  v4f a1 = *(const v4f*)(base + 4);
  v4f a2 = *(const v4f*)(base + 16);
  v4f a3 = *(const v4f*)(base + 20);
  v16h r;
#pragma unroll
  for (int i = 0; i < 4; ++i) {
    r[i]      = (_Float16)a0[i];
    r[i + 4]  = (_Float16)a1[i];
    r[i + 8]  = (_Float16)a2[i];
    r[i + 12] = (_Float16)a3[i];
  }
  return r;
}

// B fragment (32x16 f16, K x N): lane col n = l&15; lanes 0-15 hold
// K = {k0..k0+7, k0+16..k0+23}; lanes 16-31 shifted by +8. Source N x K row-major.
__device__ __forceinline__ v16h load_fragB_h(const _Float16* __restrict__ p,
                                             int ld, int n0, int k0) {
  int lane = threadIdx.x & 31;
  const _Float16* base =
      p + (size_t)(n0 + (lane & 15)) * ld + k0 + ((lane >> 4) << 3);
  union { v16h v; v8h h[2]; } u;
  u.h[0] = *(const v8h*)(base);
  u.h[1] = *(const v8h*)(base + 16);
  return u.v;
}

// ---------------------------------------------------------------------------
// Conversion / transpose pre-passes
// ---------------------------------------------------------------------------
__global__ void cvt_f32_to_f16(const float* __restrict__ in,
                               _Float16* __restrict__ out, int n) {
  int i = blockIdx.x * blockDim.x + threadIdx.x;
  if (i < n) out[i] = (_Float16)in[i];
}

// in: rows x cols (row-major fp32) -> out: cols x rows (row-major f16)
__global__ void cvt_transpose(const float* __restrict__ in,
                              _Float16* __restrict__ out, int rows, int cols) {
  int i = blockIdx.x * blockDim.x + threadIdx.x;
  if (i < rows * cols) {
    int r = i / cols, c = i % cols;
    out[(size_t)c * rows + r] = (_Float16)in[i];
  }
}

// ---------------------------------------------------------------------------
// QKV projection: (8192 x 1024) x (1024 x 3072) + bias.
// Block 128x128, 8 waves (4M x 2N), wave tile 32x64. K loop fully unrolled:
// no loop-carried phis -> accumulators stay pinned (no rotation copies), and
// the scheduler hoists future fragment loads over current WMMAs.
// ---------------------------------------------------------------------------
__global__ __launch_bounds__(256) void qkv_gemm(
    const _Float16* __restrict__ xh, const _Float16* __restrict__ wqt,
    const float* __restrict__ bqkv, _Float16* __restrict__ q,
    _Float16* __restrict__ kmat, _Float16* __restrict__ vt) {
  int wave = threadIdx.x >> 5, lane = threadIdx.x & 31;
  int row0 = blockIdx.x * 128 + (wave & 3) * 32;
  int col0 = blockIdx.y * 128 + (wave >> 2) * 64;
  v8f acc[2][4] = {};

#pragma unroll
  for (int kt = 0; kt < CH; kt += 32) {
    if ((kt & 255) == 0)
      __builtin_prefetch(xh + (size_t)(row0 + lane) * CH + kt + 256, 0, 3);
    v16h a0 = load_fragA_h(xh, CH, row0, kt);
    v16h a1 = load_fragA_h(xh, CH, row0 + 16, kt);
#pragma unroll
    for (int t = 0; t < 4; ++t) {
      v16h b = load_fragB_h(wqt, CH, col0 + t * 16, kt);
      acc[0][t] = wmma_f16(a0, b, acc[0][t]);
      acc[1][t] = wmma_f16(a1, b, acc[1][t]);
    }
  }

  int bi = row0 >> 10;  // batch index (tiles never straddle batches)
#pragma unroll
  for (int i = 0; i < 2; ++i) {
    int rb = row0 + i * 16;
#pragma unroll
    for (int t = 0; t < 4; ++t) {
      int j0 = col0 + t * 16;
      int s  = j0 >> 10;          // 0=q 1=k 2=v
      int hh = (j0 & 1023) >> 6;  // head
      int j  = j0 + (lane & 15);
      int d  = j & 63;
      float bias = bqkv[j];
      size_t bh = (size_t)(bi * HEADS + hh);
#pragma unroll
      for (int v = 0; v < 8; ++v) {
        int m = rb + v + ((lane >> 4) << 3);
        int n = m & (SEQ - 1);
        float val = acc[i][t][v] + bias;
        if (s == 0)
          q[(bh * SEQ + n) * HDIM + d] = (_Float16)(val * SCALE);
        else if (s == 1)
          kmat[(bh * SEQ + n) * HDIM + d] = (_Float16)val;
        else
          vt[(bh * HDIM + d) * SEQ + n] = (_Float16)val;
      }
    }
  }
}

// ---------------------------------------------------------------------------
// Fused scores + softmax: per (b,h), per 16-row block, S = q k^T (16x1024)
// held in registers across 4 waves; softmax; write attn fp32 once.
// ---------------------------------------------------------------------------
__global__ __launch_bounds__(128) void attn_scores(
    const _Float16* __restrict__ q, const _Float16* __restrict__ kmat,
    float* __restrict__ attn) {
  int wave = threadIdx.x >> 5, lane = threadIdx.x & 31;
  int bh = blockIdx.y;
  const _Float16* qp = q + (size_t)bh * SEQ * HDIM;
  const _Float16* kp = kmat + (size_t)bh * SEQ * HDIM;
  int row0 = blockIdx.x * 16;
  int cb = wave * 256;
  v8f acc[16] = {};
#pragma unroll
  for (int kt = 0; kt < HDIM; kt += 32) {
    v16h a = load_fragA_h(qp, HDIM, row0, kt);
#pragma unroll
    for (int t = 0; t < 16; ++t) {
      v16h b = load_fragB_h(kp, HDIM, cb + t * 16, kt);
      acc[t] = wmma_f16(a, b, acc[t]);
    }
  }

  __shared__ float red[4][16];
  int rbase = (lane >> 4) << 3;  // rows 0-7 (lanes 0-15) or 8-15 (lanes 16-31)

  // --- row max ---
  float rmax[8];
#pragma unroll
  for (int v = 0; v < 8; ++v) {
    float m = -3.0e38f;
#pragma unroll
    for (int t = 0; t < 16; ++t) m = fmaxf(m, acc[t][v]);
    for (int off = 1; off < 16; off <<= 1) m = fmaxf(m, __shfl_xor(m, off, 32));
    rmax[v] = m;
  }
  if ((lane & 15) == 0) {
#pragma unroll
    for (int v = 0; v < 8; ++v) red[wave][rbase + v] = rmax[v];
  }
  __syncthreads();
#pragma unroll
  for (int v = 0; v < 8; ++v) {
    int r = rbase + v;
    rmax[v] = fmaxf(fmaxf(red[0][r], red[1][r]), fmaxf(red[2][r], red[3][r]));
  }
  __syncthreads();

  // --- exp + row sum ---
  float rinv[8];
#pragma unroll
  for (int v = 0; v < 8; ++v) {
    float s = 0.f;
#pragma unroll
    for (int t = 0; t < 16; ++t) {
      float p = __expf(acc[t][v] - rmax[v]);
      acc[t][v] = p;
      s += p;
    }
    for (int off = 1; off < 16; off <<= 1) s += __shfl_xor(s, off, 32);
    rinv[v] = s;
  }
  if ((lane & 15) == 0) {
#pragma unroll
    for (int v = 0; v < 8; ++v) red[wave][rbase + v] = rinv[v];
  }
  __syncthreads();
#pragma unroll
  for (int v = 0; v < 8; ++v) {
    int r = rbase + v;
    rinv[v] = 1.0f / (red[0][r] + red[1][r] + red[2][r] + red[3][r]);
  }

  // --- normalize + write attn (fp32) ---
  float* outp = attn + ((size_t)bh << 20);
#pragma unroll
  for (int t = 0; t < 16; ++t) {
    int col = cb + t * 16 + (lane & 15);
#pragma unroll
    for (int v = 0; v < 8; ++v) {
      int m = row0 + rbase + v;
      outp[(size_t)m * SEQ + col] = acc[t][v] * rinv[v];
    }
  }
}

// ---------------------------------------------------------------------------
// O = attn @ V : per (b,h), M=1024, N=64, K=1024; attn fp32 -> f16 in regs.
// Wave tile 32x64, K loop fully unrolled.
// ---------------------------------------------------------------------------
__global__ __launch_bounds__(256) void attn_v(
    const float* __restrict__ attn, const _Float16* __restrict__ vt,
    _Float16* __restrict__ oh) {
  int wave = threadIdx.x >> 5, lane = threadIdx.x & 31;
  int bh = blockIdx.y;
  int bi = bh >> 4, hh = bh & 15;
  const float* ap = attn + ((size_t)bh << 20);
  const _Float16* vp = vt + (size_t)bh * HDIM * SEQ;
  int row0 = (blockIdx.x * 8 + wave) * 32;  // gridDim.x = 4
  v8f acc[2][4] = {};

#pragma unroll
  for (int kt = 0; kt < SEQ; kt += 32) {
    if ((kt & 255) == 0)
      __builtin_prefetch(ap + (size_t)(row0 + lane) * SEQ + kt + 256, 0, 3);
    v16h a0 = load_fragA_f32(ap, SEQ, row0, kt);
    v16h a1 = load_fragA_f32(ap, SEQ, row0 + 16, kt);
#pragma unroll
    for (int t = 0; t < 4; ++t) {
      v16h b = load_fragB_h(vp, SEQ, t * 16, kt);
      acc[0][t] = wmma_f16(a0, b, acc[0][t]);
      acc[1][t] = wmma_f16(a1, b, acc[1][t]);
    }
  }

#pragma unroll
  for (int i = 0; i < 2; ++i) {
#pragma unroll
    for (int t = 0; t < 4; ++t) {
      int d = t * 16 + (lane & 15);
#pragma unroll
      for (int v = 0; v < 8; ++v) {
        int m = row0 + i * 16 + v + ((lane >> 4) << 3);
        oh[((size_t)(bi * SEQ + m)) * CH + hh * HDIM + d] =
            (_Float16)acc[i][t][v];
      }
    }
  }
}

// ---------------------------------------------------------------------------
// Projection: out = O @ w_proj + b_proj  (8192x1024)x(1024x1024) -> fp32
// Block 128x128, wave tile 32x64, K loop fully unrolled.
// ---------------------------------------------------------------------------
__global__ __launch_bounds__(256) void proj_gemm(
    const _Float16* __restrict__ oh, const _Float16* __restrict__ wpt,
    const float* __restrict__ bproj, float* __restrict__ out) {
  int wave = threadIdx.x >> 5, lane = threadIdx.x & 31;
  int row0 = blockIdx.x * 128 + (wave & 3) * 32;
  int col0 = blockIdx.y * 128 + (wave >> 2) * 64;
  v8f acc[2][4] = {};

#pragma unroll
  for (int kt = 0; kt < CH; kt += 32) {
    if ((kt & 255) == 0)
      __builtin_prefetch(oh + (size_t)(row0 + lane) * CH + kt + 256, 0, 3);
    v16h a0 = load_fragA_h(oh, CH, row0, kt);
    v16h a1 = load_fragA_h(oh, CH, row0 + 16, kt);
#pragma unroll
    for (int t = 0; t < 4; ++t) {
      v16h b = load_fragB_h(wpt, CH, col0 + t * 16, kt);
      acc[0][t] = wmma_f16(a0, b, acc[0][t]);
      acc[1][t] = wmma_f16(a1, b, acc[1][t]);
    }
  }

#pragma unroll
  for (int i = 0; i < 2; ++i) {
#pragma unroll
    for (int t = 0; t < 4; ++t) {
      int c = col0 + t * 16 + (lane & 15);
      float bias = bproj[c];
#pragma unroll
      for (int v = 0; v < 8; ++v) {
        int m = row0 + i * 16 + v + ((lane >> 4) << 3);
        out[(size_t)m * CH + c] = acc[i][t][v] + bias;
      }
    }
  }
}

// ---------------------------------------------------------------------------
extern "C" void kernel_launch(void* const* d_in, const int* in_sizes, int n_in,
                              void* d_out, int out_size, void* d_ws,
                              size_t ws_size, hipStream_t stream) {
  const float* x      = (const float*)d_in[0];  // (8,1024,1024)
  const float* w_qkv  = (const float*)d_in[1];  // (1024,3072)
  const float* b_qkv  = (const float*)d_in[2];  // (3072,)
  const float* w_proj = (const float*)d_in[3];  // (1024,1024)
  const float* b_proj = (const float*)d_in[4];  // (1024,)

  float* out  = (float*)d_out;
  float* attn = out + (size_t)BATCH * SEQ * CH;  // attn after out, fp32

  const size_t EH = (size_t)BATCH * SEQ * CH;  // 8,388,608 halves per tensor

  char* ws = (char*)d_ws;
  _Float16* xh  = (_Float16*)ws;  ws += EH * 2;
  _Float16* wqt = (_Float16*)ws;  ws += (size_t)3 * CH * CH * 2;
  _Float16* wpt = (_Float16*)ws;  ws += (size_t)CH * CH * 2;
  _Float16* qf  = (_Float16*)ws;  ws += EH * 2;
  _Float16* kf  = (_Float16*)ws;  ws += EH * 2;
  _Float16* vtf = (_Float16*)ws;  ws += EH * 2;
  _Float16* ohf = (_Float16*)ws;  ws += EH * 2;

  cvt_f32_to_f16<<<(int)(EH / 256), 256, 0, stream>>>(x, xh, (int)EH);
  cvt_transpose<<<(3 * CH * CH + 255) / 256, 256, 0, stream>>>(w_qkv, wqt, CH,
                                                               3 * CH);
  cvt_transpose<<<(CH * CH + 255) / 256, 256, 0, stream>>>(w_proj, wpt, CH, CH);

  qkv_gemm<<<dim3(64, 24), 256, 0, stream>>>(xh, wqt, b_qkv, qf, kf, vtf);
  attn_scores<<<dim3(SEQ / 16, BATCH * HEADS), 128, 0, stream>>>(qf, kf, attn);
  attn_v<<<dim3(4, BATCH * HEADS), 256, 0, stream>>>(attn, vtf, ohf);
  proj_gemm<<<dim3(64, 8), 256, 0, stream>>>(ohf, wpt, b_proj, out);
}
